// BlockWrapper_47321949667976
// MI455X (gfx1250) — matrix operand order repaired and checked by reading, not compiled, not run on hardware
//
#include <hip/hip_runtime.h>
#include <hip/hip_bf16.h>

// ---------------------------------------------------------------------------
// ViT block w/ prompt-prefix attention for MI455X (gfx1250, wave32, WMMA).
// Heavy GEMMs: TDM (tensor_load_to_lds) double-buffered LDS staging (BK=64)
// feeding v_wmma_f32_16x16x32_f16 (f16 data, f32 accum), 32x32 wave tiles.
// Attention: flash-style, K staged via TDM + V staged transposed in LDS, all
// WMMA inner loops are LDS-fed and branch-free.
// ---------------------------------------------------------------------------

typedef __attribute__((ext_vector_type(16))) _Float16 v16h;
typedef __attribute__((ext_vector_type(8)))  _Float16 v8h;
typedef __attribute__((ext_vector_type(8)))  float    v8f;
typedef __attribute__((ext_vector_type(4)))  unsigned v4u;
typedef __attribute__((ext_vector_type(8)))  unsigned v8u;

#define WMMA32(a, b, c) \
  __builtin_amdgcn_wmma_f32_16x16x32_f16(false, (a), false, (b), (short)0, (c), false, false)

// Build a 16-bit A-matrix (16x32) fragment in contiguous k order (B operands use the same order):
// lane L (row = L&15): halves 0..15 = K[kb..kb+15],
// where kb = (L>>4)*16 (+ caller's K offset).
__device__ __forceinline__ v16h make_afrag(const _Float16* p, int kb) {
  v8h lo = *(const v8h*)(p + kb);
  v8h hh = *(const v8h*)(p + kb + 8);
  return __builtin_shufflevector(lo, hh, 0, 1, 2, 3, 4, 5, 6, 7, 8, 9, 10, 11, 12, 13, 14, 15);
}

// ---------------------------------------------------------------------------
// TDM: issue a 2-D tile load (rows x cols_h halves) from global into LDS.
// D# per CDNA5 ISA ch.8: group0 = {count|flags, lds_addr, global_addr, type},
// group1 = dims/strides, groups 2&3 zero (2-D tensor, fields unused).
// Wave-level op (EXEC ignored): caller must gate to a single wave.
// ---------------------------------------------------------------------------
__device__ __forceinline__ void tdm_load_2d(unsigned lds_byte_off,
                                            const _Float16* gptr,
                                            unsigned tile_rows,
                                            unsigned tile_cols_h,
                                            unsigned row_stride_h,
                                            unsigned tensor_rows) {
  unsigned long long ga = (unsigned long long)(size_t)gptr;
  v4u g0;
  g0[0] = 1u;                                   // count=1, user D#, no gather
  g0[1] = lds_byte_off;                         // lds_addr
  g0[2] = (unsigned)ga;                         // global_addr[31:0]
  g0[3] = (unsigned)((ga >> 32) & 0x01FFFFFFull) | 0x80000000u;  // addr[56:32] | type=2
  v8u g1;
  g1[0] = (1u << 16);                           // data_size=1 (2 bytes/elem), no flags
  g1[1] = (row_stride_h & 0xFFFFu) << 16;       // tensor_dim0[15:0]
  g1[2] = ((row_stride_h >> 16) & 0xFFFFu) | ((tensor_rows & 0xFFFFu) << 16);
  g1[3] = ((tensor_rows >> 16) & 0xFFFFu) | (tile_cols_h << 16);  // dim1 hi | tile_dim0
  g1[4] = tile_rows & 0xFFFFu;                  // tile_dim1 (tile_dim2 = 0)
  g1[5] = row_stride_h;                         // tensor_dim0_stride[31:0]
  g1[6] = 0u;
  g1[7] = 0u;
  v4u gz = {0u, 0u, 0u, 0u};
  asm volatile("tensor_load_to_lds %0, %1, %2, %3"
               :
               : "s"(g0), "s"(g1), "s"(gz), "s"(gz)
               : "memory");
}

// ---------------------------------------------------------------------------
// f32 -> f16 conversion (weights)
// ---------------------------------------------------------------------------
__global__ void cvt_f16_kernel(const float* __restrict__ in, _Float16* __restrict__ out, int n) {
  int i = blockIdx.x * blockDim.x + threadIdx.x;
  if (i < n) out[i] = (_Float16)in[i];
}

// ---------------------------------------------------------------------------
// LayerNorm (C = 768) -> f16 output. One block (256 threads) per row.
// ---------------------------------------------------------------------------
__global__ __launch_bounds__(256) void ln_f16_kernel(
    const float* __restrict__ x, const float* __restrict__ w,
    const float* __restrict__ bcoef, _Float16* __restrict__ out) {
  const int row = blockIdx.x;
  const int tid = threadIdx.x;
  const float* xr = x + (size_t)row * 768;

  float v[3], s = 0.0f, sq = 0.0f;
#pragma unroll
  for (int i = 0; i < 3; ++i) {
    v[i] = xr[tid + i * 256];
    s += v[i];
    sq += v[i] * v[i];
  }
  __shared__ float red[512];
  red[tid] = s;
  red[256 + tid] = sq;
  __syncthreads();
  for (int st = 128; st > 0; st >>= 1) {
    if (tid < st) {
      red[tid] += red[tid + st];
      red[256 + tid] += red[256 + tid + st];
    }
    __syncthreads();
  }
  const float mu  = red[0] * (1.0f / 768.0f);
  const float var = red[256] * (1.0f / 768.0f) - mu * mu;
  const float inv = rsqrtf(var + 1e-6f);
#pragma unroll
  for (int i = 0; i < 3; ++i) {
    int c = tid + i * 256;
    out[(size_t)row * 768 + c] = (_Float16)((v[i] - mu) * inv * w[c] + bcoef[c]);
  }
}

// ---------------------------------------------------------------------------
// WMMA GEMM with TDM double-buffered LDS staging.
// out[M,N] = epilogue(A[M,K] @ W[N,K]^T + bias [+ res]).
// Block tile 64(M) x 128(N), BK = 64; 8 waves in 2(M) x 4(N); wave tile 32x32
// (8 WMMAs / K-step). Requires M%64==0, N%128==0, K%64==0 (true here).
// ---------------------------------------------------------------------------
template <bool BIAS, bool DO_GELU, bool RES, bool OUT16>
__device__ __forceinline__ void store_tile(
    v8f acc, int mbase, int col, int N,
    const float* __restrict__ bias, const float* __restrict__ res,
    float* __restrict__ outF, _Float16* __restrict__ outH) {
  const float bv = BIAS ? bias[col] : 0.0f;
#pragma unroll
  for (int r = 0; r < 8; ++r) {
    float v = acc[r] + bv;
    if (DO_GELU) v = 0.5f * v * (1.0f + erff(v * 0.70710678118f));
    size_t idx = (size_t)(mbase + r) * N + col;
    if (RES) v += res[idx];
    if (OUT16) outH[idx] = (_Float16)v;
    else       outF[idx] = v;
  }
}

template <bool BIAS, bool DO_GELU, bool RES, bool OUT16>
__global__ __launch_bounds__(256) void gemm_wmma_kernel(
    const _Float16* __restrict__ A, const _Float16* __restrict__ W,
    const float* __restrict__ bias, const float* __restrict__ res,
    float* __restrict__ outF, _Float16* __restrict__ outH,
    int M, int N, int K) {
  __shared__ _Float16 aTile[2][64 * 64];    //  8 KB per buffer
  __shared__ _Float16 bTile[2][128 * 64];   // 16 KB per buffer

  const int lane = threadIdx.x & 31;
  const int wave = threadIdx.x >> 5;
  const int lo16 = lane & 15;
  const int hig  = lane >> 4;
  const int wm = wave & 1;   // 2 M-waves of 32 rows
  const int wn = wave >> 1;  // 4 N-waves of 32 cols
  const int m0 = blockIdx.y * 64;
  const int n0 = blockIdx.x * 128;
  const bool issuer = (wave == 0);

  const unsigned aOff0 = (unsigned)(size_t)&aTile[0][0];
  const unsigned aOff1 = (unsigned)(size_t)&aTile[1][0];
  const unsigned bOff0 = (unsigned)(size_t)&bTile[0][0];
  const unsigned bOff1 = (unsigned)(size_t)&bTile[1][0];

  // Prologue: DMA buffer 0.
  if (issuer) {
    tdm_load_2d(aOff0, A + (size_t)m0 * K, 64, 64, (unsigned)K, (unsigned)M);
    tdm_load_2d(bOff0, W + (size_t)n0 * K, 128, 64, (unsigned)K, (unsigned)N);
    __builtin_amdgcn_s_wait_tensorcnt(0);
  }
  __syncthreads();

  v8f acc[2][2];
#pragma unroll
  for (int i = 0; i < 2; ++i)
#pragma unroll
    for (int j = 0; j < 2; ++j) acc[i][j] = (v8f){0, 0, 0, 0, 0, 0, 0, 0};

  const int nsteps = K >> 6;
  for (int s = 0; s < nsteps; ++s) {
    const int cur = s & 1;
    if (issuer && (s + 1) < nsteps) {  // prefetch next K-slab into the other buffer
      const int kk = (s + 1) * 64;
      tdm_load_2d(cur ? aOff0 : aOff1, A + (size_t)m0 * K + kk, 64, 64, (unsigned)K, (unsigned)M);
      tdm_load_2d(cur ? bOff0 : bOff1, W + (size_t)n0 * K + kk, 128, 64, (unsigned)K, (unsigned)N);
    }
    const _Float16* at = &aTile[cur][0];
    const _Float16* bt = &bTile[cur][0];
#pragma unroll
    for (int kc = 0; kc < 2; ++kc) {  // two 32-wide sub-steps of the 64-wide slab
      v16h af[2], bf[2];
#pragma unroll
      for (int i = 0; i < 2; ++i)
        af[i] = make_afrag(at + (wm * 32 + i * 16 + lo16) * 64, kc * 32 + (hig << 4));
#pragma unroll
      for (int j = 0; j < 2; ++j)
        bf[j] = *(const v16h*)(bt + (wn * 32 + j * 16 + lo16) * 64 + kc * 32 + (hig << 4));
#pragma unroll
      for (int i = 0; i < 2; ++i)
#pragma unroll
        for (int j = 0; j < 2; ++j) acc[i][j] = WMMA32(af[i], bf[j], acc[i][j]);
    }

    __syncthreads();                                   // all waves done reading buf[cur]
    if (issuer && (s + 1) < nsteps) __builtin_amdgcn_s_wait_tensorcnt(0);
    __syncthreads();                                   // next buffer fully in LDS
  }

#pragma unroll
  for (int i = 0; i < 2; ++i) {
    const int mbase = m0 + wm * 32 + i * 16 + (hig << 3);
#pragma unroll
    for (int j = 0; j < 2; ++j) {
      const int col = n0 + wn * 32 + j * 16 + lo16;
      store_tile<BIAS, DO_GELU, RES, OUT16>(acc[i][j], mbase, col, N, bias, res, outF, outH);
    }
  }
}

// ---------------------------------------------------------------------------
// Flash attention with prompt prefix. One block per (h, b); 8 waves; each wave
// owns 16-query tiles t = wave, wave+8 (13 tiles cover N=197).
// K is staged in LDS (prompt rows converted, qkv rows via TDM); V is staged
// TRANSPOSED in LDS so PV B-fragments are contiguous 32B ds reads.
// LDS: K 217x64 (27.1KB) + Vt 64x224 (28KB) + P slabs (8KB) = 63.1 KB.
// ---------------------------------------------------------------------------
__global__ __launch_bounds__(256) void attn_flash_kernel(
    const _Float16* __restrict__ qkv, const float* __restrict__ prompt,
    _Float16* __restrict__ attO) {
  const int h    = blockIdx.x;
  const int b    = blockIdx.y;
  const int tid  = threadIdx.x;
  const int wave = tid >> 5;
  const int lane = tid & 31;
  const int lo16 = lane & 15;
  const int hig  = lane >> 4;

  __shared__ _Float16 Klds[217 * 64];     // [key][d]
  __shared__ _Float16 Vtlds[64 * 224];    // [d][key] (transposed)
  __shared__ _Float16 pslab[8][16 * 32];  // per-wave P staging

  // --- TDM: K rows 20..216 straight out of the qkv buffer (197x64, stride 2304)
  if (wave == 0) {
    tdm_load_2d((unsigned)(size_t)&Klds[20 * 64],
                qkv + (size_t)(b * 197) * 2304 + 768 + h * 64,
                197, 64, 2304, 197);
  }
  // --- prompt K rows (f32 -> f16) ---
  for (int idx = tid; idx < 20 * 64; idx += 256) {
    int r = idx >> 6, d = idx & 63;
    Klds[r * 64 + d] = (_Float16)prompt[((((size_t)b * 20 + r) * 2 + 0) * 12 + h) * 64 + d];
  }
  // --- V transposed into LDS (keys 217..223 zeroed; P is 0 there) ---
  for (int idx = tid; idx < 224 * 64; idx += 256) {
    int key = idx >> 6, d = idx & 63;
    float v;
    if (key < 20)       v = prompt[((((size_t)b * 20 + key) * 2 + 1) * 12 + h) * 64 + d];
    else if (key < 217) v = (float)qkv[(size_t)(b * 197 + key - 20) * 2304 + 1536 + h * 64 + d];
    else                v = 0.0f;
    Vtlds[d * 224 + key] = (_Float16)v;
  }
  if (wave == 0) __builtin_amdgcn_s_wait_tensorcnt(0);
  __syncthreads();
  // Waves are independent from here (no further block barriers).

  _Float16* slab = &pslab[wave][0];
  const float scale = 0.125f;

  for (int t = wave; t < 13; t += 8) {
    const int nbase = t * 16;
    int qrow = nbase + lo16;
    if (qrow > 196) qrow = 196;  // clamp pad rows (never stored)
    const _Float16* qp = qkv + (size_t)(b * 197 + qrow) * 2304 + h * 64;
    v16h qa0 = make_afrag(qp, (hig << 4));
    v16h qa1 = make_afrag(qp, 32 + (hig << 4));

    float mrow[8], lsum[8];
    v8f o0 = {0, 0, 0, 0, 0, 0, 0, 0}, o1 = o0, o2 = o0, o3 = o0;
#pragma unroll
    for (int r = 0; r < 8; ++r) { mrow[r] = -3.0e38f; lsum[r] = 0.0f; }

    const int key0 = lo16;                       // within 32-key step, tile 0
    for (int ks = 0; ks < 7; ++ks) {             // 7 * 32 = 224 >= 217 keys
      v8f s0 = {0, 0, 0, 0, 0, 0, 0, 0}, s1 = s0;
      const int k0 = ks * 32 + key0;             // always <= 207
      int k1 = ks * 32 + 16 + lo16;
      const bool dead1 = (k1 >= 217);
      if (k1 > 216) k1 = 216;                    // clamp LDS row; column masked below
#pragma unroll
      for (int ds = 0; ds < 2; ++ds) {
        const int dbase = ds * 32 + (hig << 4);
        v16h bk0 = *(const v16h*)(&Klds[k0 * 64 + dbase]);
        v16h bk1 = *(const v16h*)(&Klds[k1 * 64 + dbase]);
        v16h qa = ds ? qa1 : qa0;
        s0 = WMMA32(qa, bk0, s0);
        s1 = WMMA32(qa, bk1, s1);
      }
      float mloc[8], ts[8];
#pragma unroll
      for (int r = 0; r < 8; ++r) {
        float a0 = s0[r] * scale;
        float a1 = dead1 ? -3.0e38f : s1[r] * scale;
        s0[r] = a0;
        s1[r] = a1;
        mloc[r] = fmaxf(a0, a1);
      }
#pragma unroll
      for (int off = 1; off < 16; off <<= 1) {
#pragma unroll
        for (int r = 0; r < 8; ++r)
          mloc[r] = fmaxf(mloc[r], __shfl_xor(mloc[r], off, 32));
      }
#pragma unroll
      for (int r = 0; r < 8; ++r) {
        float mn  = fmaxf(mrow[r], mloc[r]);
        float fac = __expf(mrow[r] - mn);
        mrow[r] = mn;
        lsum[r] *= fac;
        o0[r] *= fac; o1[r] *= fac; o2[r] *= fac; o3[r] *= fac;
        float e0 = __expf(s0[r] - mn);
        float e1 = __expf(s1[r] - mn);
        slab[((hig << 3) + r) * 32 + lo16]      = (_Float16)e0;
        slab[((hig << 3) + r) * 32 + 16 + lo16] = (_Float16)e1;
        ts[r] = e0 + e1;
      }
#pragma unroll
      for (int off = 1; off < 16; off <<= 1) {
#pragma unroll
        for (int r = 0; r < 8; ++r) ts[r] += __shfl_xor(ts[r], off, 32);
      }
#pragma unroll
      for (int r = 0; r < 8; ++r) lsum[r] += ts[r];

      asm volatile("s_wait_dscnt 0" ::: "memory");  // P slab stores visible to A-frag loads
      v16h pa = make_afrag(slab + lo16 * 32, (hig << 4));
      const int kbase = ks * 32 + (hig << 4);
      v16h vb0 = *(const v16h*)(&Vtlds[( 0 + lo16) * 224 + kbase]);
      v16h vb1 = *(const v16h*)(&Vtlds[(16 + lo16) * 224 + kbase]);
      v16h vb2 = *(const v16h*)(&Vtlds[(32 + lo16) * 224 + kbase]);
      v16h vb3 = *(const v16h*)(&Vtlds[(48 + lo16) * 224 + kbase]);
      o0 = WMMA32(pa, vb0, o0);
      o1 = WMMA32(pa, vb1, o1);
      o2 = WMMA32(pa, vb2, o2);
      o3 = WMMA32(pa, vb3, o3);
    }

#pragma unroll
    for (int r = 0; r < 8; ++r) {
      int m = nbase + (hig << 3) + r;
      if (m < 197) {
        float inv = 1.0f / lsum[r];
        _Float16* op = attO + (size_t)(b * 197 + m) * 768 + h * 64;
        op[ 0 + lo16] = (_Float16)(o0[r] * inv);
        op[16 + lo16] = (_Float16)(o1[r] * inv);
        op[32 + lo16] = (_Float16)(o2[r] * inv);
        op[48 + lo16] = (_Float16)(o3[r] * inv);
      }
    }
  }
}

// ---------------------------------------------------------------------------
// Host launch
// ---------------------------------------------------------------------------
extern "C" void kernel_launch(void* const* d_in, const int* in_sizes, int n_in,
                              void* d_out, int out_size, void* d_ws, size_t ws_size,
                              hipStream_t stream) {
  const float* x      = (const float*)d_in[0];
  const float* prompt = (const float*)d_in[1];
  const float* ln1_w  = (const float*)d_in[2];
  const float* ln1_b  = (const float*)d_in[3];
  const float* qkv_w  = (const float*)d_in[4];
  const float* qkv_b  = (const float*)d_in[5];
  const float* proj_w = (const float*)d_in[6];
  const float* proj_b = (const float*)d_in[7];
  const float* ln2_w  = (const float*)d_in[8];
  const float* ln2_b  = (const float*)d_in[9];
  const float* fc1_w  = (const float*)d_in[10];
  const float* fc1_b  = (const float*)d_in[11];
  const float* fc2_w  = (const float*)d_in[12];
  const float* fc2_b  = (const float*)d_in[13];
  float* out = (float*)d_out;

  const int R = 64 * 197;  // 12608 rows

  // Workspace carve (all offsets 32B-aligned; total ~227 MB)
  char* ws = (char*)d_ws;
  size_t off = 0;
  _Float16* qkvw16 = (_Float16*)(ws + off); off += (size_t)2304 * 768 * 2;
  _Float16* projw16 = (_Float16*)(ws + off); off += (size_t)768 * 768 * 2;
  _Float16* fc1w16 = (_Float16*)(ws + off); off += (size_t)3072 * 768 * 2;
  _Float16* fc2w16 = (_Float16*)(ws + off); off += (size_t)768 * 3072 * 2;
  _Float16* h16    = (_Float16*)(ws + off); off += (size_t)R * 768 * 2;
  _Float16* qkv16  = (_Float16*)(ws + off); off += (size_t)R * 2304 * 2;
  _Float16* atto16 = (_Float16*)(ws + off); off += (size_t)R * 768 * 2;
  _Float16* m116   = (_Float16*)(ws + off); off += (size_t)R * 3072 * 2;
  float*    x1     = (float*)(ws + off);    off += (size_t)R * 768 * 4;

  // 1) convert weights to f16
  {
    int n;
    n = 2304 * 768; cvt_f16_kernel<<<(n + 255) / 256, 256, 0, stream>>>(qkv_w, qkvw16, n);
    n = 768 * 768;  cvt_f16_kernel<<<(n + 255) / 256, 256, 0, stream>>>(proj_w, projw16, n);
    n = 3072 * 768; cvt_f16_kernel<<<(n + 255) / 256, 256, 0, stream>>>(fc1_w, fc1w16, n);
    n = 768 * 3072; cvt_f16_kernel<<<(n + 255) / 256, 256, 0, stream>>>(fc2_w, fc2w16, n);
  }

  // 2) LN1 -> h16
  ln_f16_kernel<<<R, 256, 0, stream>>>(x, ln1_w, ln1_b, h16);

  // 3) QKV = h16 @ qkv_w^T + b  (f16 out)
  gemm_wmma_kernel<true, false, false, true><<<dim3(2304 / 128, R / 64), 256, 0, stream>>>(
      h16, qkvw16, qkv_b, nullptr, nullptr, qkv16, R, 2304, 768);

  // 4) flash attention with prompt prefix -> atto16
  attn_flash_kernel<<<dim3(12, 64), 256, 0, stream>>>(qkv16, prompt, atto16);

  // 5) x1 = x + atto16 @ proj_w^T + b  (f32 out)
  gemm_wmma_kernel<true, false, true, false><<<dim3(768 / 128, R / 64), 256, 0, stream>>>(
      atto16, projw16, proj_b, x, x1, nullptr, R, 768, 768);

  // 6) LN2 -> h16 (reused)
  ln_f16_kernel<<<R, 256, 0, stream>>>(x1, ln2_w, ln2_b, h16);

  // 7) m1 = gelu(h16 @ fc1_w^T + b)  (f16 out)
  gemm_wmma_kernel<true, true, false, true><<<dim3(3072 / 128, R / 64), 256, 0, stream>>>(
      h16, fc1w16, fc1_b, nullptr, nullptr, m116, R, 3072, 768);

  // 8) out = x1 + m1 @ fc2_w^T + b  (f32 out)
  gemm_wmma_kernel<true, false, true, false><<<dim3(768 / 128, R / 64), 256, 0, stream>>>(
      m116, fc2w16, fc2_b, x1, out, nullptr, R, 768, 3072);
}
